// OSS_80444737454936
// MI455X (gfx1250) — matrix operand order repaired and checked by reading, not compiled, not run on hardware
//
#include <hip/hip_runtime.h>
#include <hip/hip_bf16.h>

// ---------------- problem constants ----------------
#define Bsz   4
#define Cch   32
#define Hh    64
#define Ww    64
#define Lseq  4096            // H*W
#define Mrows (Bsz * Lseq)    // 16384
#define DM1   128             // 4*C
#define DI1   256             // 8*C
#define NSTATE 16
#define DTR1  8
#define XDBLN 48              // 40 padded to 48 for WMMA

typedef __attribute__((ext_vector_type(16))) __bf16 v16bf;
typedef __attribute__((ext_vector_type(8)))  float  v8f;

// ---------------------------------------------------------------------------
// WMMA bf16 GEMM:  C(MxN) = A(MxK) * B(NxK)^T   (row-major A and B)
// One wave computes a 16 x (16*NT) strip of C: the A fragment is loaded once
// per K-step and reused for NT WMMAs (cuts A-operand traffic NT x, which is
// the dominant byte stream for these skinny-N GEMMs).
// Fragment layouts used by this code (wave32, contiguous K order):
//   A 16x32 bf16: lane = M row; element e holds K = e + 16*(lane>=16)
//                 -> one contiguous 32B run per lane
//   B 32x16 bf16: lane = N col; element e holds K = e + 16*(lane>=16)
//   C 16x16 f32 : VGPR r -> M = r + 8*(lane>=16), N = lane&15
// ---------------------------------------------------------------------------
template <int NT>
__global__ void wmma_gemm_bf16(const __bf16* __restrict__ A,
                               const __bf16* __restrict__ B,
                               float* __restrict__ C,
                               int M, int N, int K) {
  const int lane = threadIdx.x & 31;
  const int m0 = blockIdx.x * 16;
  const int n0 = blockIdx.y * (16 * NT);
  const int ml = lane & 15;
  const int kh = lane >> 4;           // which K half this lane carries

  const __bf16* Arow = A + (size_t)(m0 + ml) * K;
  const __bf16* Brow[NT];
#pragma unroll
  for (int t = 0; t < NT; ++t)
    Brow[t] = B + (size_t)(n0 + t * 16 + ml) * K;

  v8f c[NT];
#pragma unroll
  for (int t = 0; t < NT; ++t) c[t] = v8f{};

  for (int k0 = 0; k0 < K; k0 += 32) {
    v16bf a;
#pragma unroll
    for (int j = 0; j < 8; ++j) {
      const int kb = 2 * j + 16 * kh;
      a[2 * j]     = Arow[k0 + kb];
      a[2 * j + 1] = Arow[k0 + kb + 1];
    }
#pragma unroll
    for (int t = 0; t < NT; ++t) {
      v16bf b = *(const v16bf*)(Brow[t] + k0 + 16 * kh);
      c[t] = __builtin_amdgcn_wmma_f32_16x16x32_bf16(false, a, false, b,
                                                     (short)0, c[t], false,
                                                     false);
    }
  }
#pragma unroll
  for (int t = 0; t < NT; ++t)
#pragma unroll
    for (int r = 0; r < 8; ++r)
      C[(size_t)(m0 + r + 8 * kh) * N + n0 + t * 16 + ml] = c[t][r];
}

// ---------------------------------------------------------------------------
// Weight prep: fp32 -> bf16 copies (+ zero-pad Wx rows 40..47), A1 = -exp(Alog)
// ---------------------------------------------------------------------------
__global__ void prep_weights(const float* __restrict__ Win,
                             const float* __restrict__ Wout,
                             const float* __restrict__ Wx,
                             const float* __restrict__ Alog,
                             __bf16* __restrict__ Winbf,
                             __bf16* __restrict__ Woutbf,
                             __bf16* __restrict__ Wxpadbf,
                             float* __restrict__ A1) {
  const int idx = blockIdx.x * 256 + threadIdx.x;     // 0 .. 65535
  if (idx < 512 * DM1)  Winbf[idx]  = (__bf16)Win[idx];
  if (idx < DM1 * DI1)  Woutbf[idx] = (__bf16)Wout[idx];
  if (idx < XDBLN * DI1) {
    const int j = idx >> 8;                            // row of padded Wx (48x256)
    Wxpadbf[idx] = (__bf16)((j < 40) ? Wx[idx] : 0.f);
  }
  if (idx < DI1 * NSTATE) A1[idx] = -__expf(Alog[idx]);
}

// ---------------------------------------------------------------------------
// Build the 4-direction line-sequence matrix fop (Mrows x 128) as bf16.
// col = dir*32 + c ; dir0 = hf, dir1 = hb, dir2 = wf, dir3 = wb
// ---------------------------------------------------------------------------
__global__ void build_fop(const float* __restrict__ fo1,
                          __bf16* __restrict__ fopbf) {
  const int idx = blockIdx.x * 256 + threadIdx.x;     // Mrows*128
  const int col = idx & 127;
  const int m   = idx >> 7;
  const int b   = m >> 12;
  const int l   = m & (Lseq - 1);
  const int dir = col >> 5;
  const int c   = col & 31;
  int hw;
  if (dir == 0)       hw = l;
  else if (dir == 1)  hw = (Lseq - 1) - l;
  else {
    const int i = (dir == 2) ? l : (Lseq - 1) - l;
    hw = (i & 63) * 64 + (i >> 6);                    // transpose(H,W) walk
  }
  fopbf[idx] = (__bf16)fo1[((size_t)(b * Cch + c)) * Lseq + hw];
}

// ---------------------------------------------------------------------------
// Causal depthwise conv (K=4) over xz[:, 0:256] + SiLU -> xa (f32 + bf16)
// ---------------------------------------------------------------------------
__global__ void conv_silu(const float* __restrict__ xz,
                          const float* __restrict__ convw,
                          const float* __restrict__ convb,
                          float* __restrict__ xa,
                          __bf16* __restrict__ xabf) {
  const int idx = blockIdx.x * 256 + threadIdx.x;     // Mrows*256
  const int d = idx & 255;
  const int m = idx >> 8;
  const int b = m >> 12;
  const int l = m & (Lseq - 1);
  float s = convb[d];
#pragma unroll
  for (int k = 0; k < 4; ++k) {
    const int li = l - 3 + k;
    if (li >= 0)
      s += convw[d * 4 + k] * xz[((size_t)(b * Lseq + li)) * 512 + d];
  }
  const float v = s / (1.f + __expf(-s));
  xa[idx]  = v;
  xabf[idx] = (__bf16)v;
}

// ---------------------------------------------------------------------------
// dt = softplus(xdbl[:, :8] @ Wdt.T + dtb)
// ---------------------------------------------------------------------------
__global__ void dt_softplus(const float* __restrict__ xdbl,
                            const float* __restrict__ Wdt,
                            const float* __restrict__ dtb,
                            float* __restrict__ dt) {
  const int idx = blockIdx.x * 256 + threadIdx.x;     // Mrows*256
  const int d = idx & 255;
  const int m = idx >> 8;
  float acc = dtb[d];
#pragma unroll
  for (int r = 0; r < DTR1; ++r)
    acc += xdbl[(size_t)m * XDBLN + r] * Wdt[d * DTR1 + r];
  dt[idx] = (acc > 20.f) ? acc : log1pf(__expf(acc));
}

// ---------------------------------------------------------------------------
// Selective scan + output gating. One thread per (batch, channel): 16 states
// in registers, serial over L=4096.  Emits yg = (y + D*xa) * silu(z) as bf16.
// ---------------------------------------------------------------------------
__global__ void scan_gate(const float* __restrict__ dt,
                          const float* __restrict__ xa,
                          const float* __restrict__ xdbl,
                          const float* __restrict__ A1,
                          const float* __restrict__ Dvec,
                          const float* __restrict__ xz,
                          __bf16* __restrict__ ygbf) {
  const int t = blockIdx.x * 256 + threadIdx.x;       // 1024 threads
  const int b = t >> 8;
  const int d = t & 255;
  float Al[NSTATE], h[NSTATE];
#pragma unroll
  for (int n = 0; n < NSTATE; ++n) { Al[n] = A1[d * NSTATE + n]; h[n] = 0.f; }
  const float Dd = Dvec[d];
  for (int l = 0; l < Lseq; ++l) {
    const size_t m = (size_t)b * Lseq + l;
    const float dtv = dt[m * 256 + d];
    const float xav = xa[m * 256 + d];
    const float du  = dtv * xav;
    float y = 0.f;
#pragma unroll
    for (int n = 0; n < NSTATE; ++n) {
      const float dA = __expf(dtv * Al[n]);
      const float Bn = xdbl[m * XDBLN + DTR1 + n];
      const float Cn = xdbl[m * XDBLN + DTR1 + NSTATE + n];
      h[n] = dA * h[n] + du * Bn;
      y += h[n] * Cn;
    }
    y += Dd * xav;
    const float zv = xz[m * 512 + 256 + d];
    const float g  = zv / (1.f + __expf(-zv));
    ygbf[m * 256 + d] = (__bf16)(y * g);
  }
}

// ---------------------------------------------------------------------------
// Sum the 4 directions, multiply by fo2 -> residual (B,C,H,W layout)
// ---------------------------------------------------------------------------
__global__ void sum_residual(const float* __restrict__ out1,
                             const float* __restrict__ fo2,
                             float* __restrict__ residual) {
  const int idx = blockIdx.x * 256 + threadIdx.x;     // B*C*Lseq
  const int hw = idx & (Lseq - 1);
  const int bc = idx >> 12;
  const int b  = bc >> 5;
  const int c  = bc & 31;
  const size_t m = (size_t)b * Lseq + hw;
  float s = 0.f;
#pragma unroll
  for (int dir = 0; dir < 4; ++dir)
    s += out1[m * DM1 + dir * Cch + c];
  residual[idx] = s * fo2[idx];
}

// ---------------------------------------------------------------------------
// Per-(b,c) mean of residual over H*W
// ---------------------------------------------------------------------------
__global__ void foc_mean(const float* __restrict__ residual,
                         float* __restrict__ focm) {
  __shared__ float sm[256];
  const int bc  = blockIdx.x;                         // 128 blocks
  const int tid = threadIdx.x;
  float s = 0.f;
  for (int i = tid; i < Lseq; i += 256)
    s += residual[(size_t)bc * Lseq + i];
  sm[tid] = s;
  __syncthreads();
  for (int st = 128; st > 0; st >>= 1) {
    if (tid < st) sm[tid] += sm[tid + st];
    __syncthreads();
  }
  if (tid == 0) focm[bc] = sm[0] * (1.f / (float)Lseq);
}

// ---------------------------------------------------------------------------
// Tiny second Mamba: L=32, d_model=2, d_inner=4, dt_rank=1. 1 thread / batch.
// focf[b,c] = sum over the 2 output features.
// ---------------------------------------------------------------------------
__global__ void mamba2_kernel(const float* __restrict__ focm,
                              const float* __restrict__ Win2,
                              const float* __restrict__ convw2,
                              const float* __restrict__ convb2,
                              const float* __restrict__ Wx2,
                              const float* __restrict__ Wdt2,
                              const float* __restrict__ dtb2,
                              const float* __restrict__ Alog2,
                              const float* __restrict__ D2,
                              const float* __restrict__ Wout2,
                              float* __restrict__ focf) {
  const int b = threadIdx.x;
  if (b >= Bsz) return;
  float xc2[32][4], z2[32][4], xa2[32][4];
  for (int l = 0; l < 32; ++l) {
    const float x0 = focm[b * 32 + l];
    const float x1 = focm[b * 32 + 31 - l];
    for (int j = 0; j < 8; ++j) {
      const float v = x0 * Win2[j * 2 + 0] + x1 * Win2[j * 2 + 1];
      if (j < 4) xc2[l][j] = v; else z2[l][j - 4] = v;
    }
  }
  for (int l = 0; l < 32; ++l)
    for (int d = 0; d < 4; ++d) {
      float s = convb2[d];
      for (int k = 0; k < 4; ++k) {
        const int li = l - 3 + k;
        if (li >= 0) s += convw2[d * 4 + k] * xc2[li][d];
      }
      xa2[l][d] = s / (1.f + __expf(-s));
    }
  float A2[4][NSTATE], h[4][NSTATE];
  for (int d = 0; d < 4; ++d)
    for (int n = 0; n < NSTATE; ++n) {
      A2[d][n] = -__expf(Alog2[d * NSTATE + n]);
      h[d][n] = 0.f;
    }
  for (int l = 0; l < 32; ++l) {
    float dtin = 0.f;
    for (int d = 0; d < 4; ++d) dtin += xa2[l][d] * Wx2[0 * 4 + d];
    float Bn[NSTATE], Cn[NSTATE];
    for (int n = 0; n < NSTATE; ++n) {
      float bs = 0.f, cs = 0.f;
      for (int d = 0; d < 4; ++d) {
        bs += xa2[l][d] * Wx2[(1 + n) * 4 + d];
        cs += xa2[l][d] * Wx2[(1 + NSTATE + n) * 4 + d];
      }
      Bn[n] = bs; Cn[n] = cs;
    }
    float o0 = 0.f, o1 = 0.f;
    for (int d = 0; d < 4; ++d) {
      float dtv = dtin * Wdt2[d] + dtb2[d];
      dtv = (dtv > 20.f) ? dtv : log1pf(__expf(dtv));
      const float du = dtv * xa2[l][d];
      float y = 0.f;
      for (int n = 0; n < NSTATE; ++n) {
        const float dA = __expf(dtv * A2[d][n]);
        h[d][n] = dA * h[d][n] + du * Bn[n];
        y += h[d][n] * Cn[n];
      }
      y += D2[d] * xa2[l][d];
      const float zv = z2[l][d];
      const float g  = zv / (1.f + __expf(-zv));
      const float yg = y * g;
      o0 += yg * Wout2[0 * 4 + d];
      o1 += yg * Wout2[1 * 4 + d];
    }
    focf[b * 32 + l] = o0 + o1;
  }
}

// ---------------------------------------------------------------------------
// out = residual * (1 + focf[b,c])
// ---------------------------------------------------------------------------
__global__ void final_kernel(const float* __restrict__ residual,
                             const float* __restrict__ focf,
                             float* __restrict__ out) {
  const int idx = blockIdx.x * 256 + threadIdx.x;     // B*C*Lseq
  out[idx] = residual[idx] * (1.f + focf[idx >> 12]);
}

// ---------------------------------------------------------------------------
extern "C" void kernel_launch(void* const* d_in, const int* in_sizes, int n_in,
                              void* d_out, int out_size, void* d_ws, size_t ws_size,
                              hipStream_t stream) {
  (void)in_sizes; (void)n_in; (void)out_size; (void)ws_size;
  const float* fo1     = (const float*)d_in[0];
  const float* fo2     = (const float*)d_in[1];
  const float* m1_Win  = (const float*)d_in[2];
  const float* m1_convw= (const float*)d_in[3];
  const float* m1_convb= (const float*)d_in[4];
  const float* m1_Wx   = (const float*)d_in[5];
  const float* m1_Wdt  = (const float*)d_in[6];
  const float* m1_dtb  = (const float*)d_in[7];
  const float* m1_Alog = (const float*)d_in[8];
  const float* m1_D    = (const float*)d_in[9];
  const float* m1_Wout = (const float*)d_in[10];
  const float* m2_Win  = (const float*)d_in[11];
  const float* m2_convw= (const float*)d_in[12];
  const float* m2_convb= (const float*)d_in[13];
  const float* m2_Wx   = (const float*)d_in[14];
  const float* m2_Wdt  = (const float*)d_in[15];
  const float* m2_dtb  = (const float*)d_in[16];
  const float* m2_Alog = (const float*)d_in[17];
  const float* m2_D    = (const float*)d_in[18];
  const float* m2_Wout = (const float*)d_in[19];
  float* out = (float*)d_out;

  // ---- workspace carve (256B aligned) ----
  char* p = (char*)d_ws;
  auto take = [&](size_t bytes) -> char* {
    char* r = p;
    p += (bytes + 255) & ~(size_t)255;
    return r;
  };
  __bf16* fopbf   = (__bf16*)take((size_t)Mrows * DM1 * 2);
  __bf16* Winbf   = (__bf16*)take((size_t)512 * DM1 * 2);
  __bf16* Wxpadbf = (__bf16*)take((size_t)XDBLN * DI1 * 2);
  __bf16* Woutbf  = (__bf16*)take((size_t)DM1 * DI1 * 2);
  float*  A1      = (float*) take((size_t)DI1 * NSTATE * 4);
  float*  xz      = (float*) take((size_t)Mrows * 512 * 4);
  float*  xa      = (float*) take((size_t)Mrows * DI1 * 4);
  __bf16* xabf    = (__bf16*)take((size_t)Mrows * DI1 * 2);
  float*  xdbl    = (float*) take((size_t)Mrows * XDBLN * 4);
  float*  dt      = (float*) take((size_t)Mrows * DI1 * 4);
  __bf16* ygbf    = (__bf16*)take((size_t)Mrows * DI1 * 2);
  float*  out1    = (float*) take((size_t)Mrows * DM1 * 4);
  float*  residual= (float*) take((size_t)Bsz * Cch * Lseq * 4);
  float*  focm    = (float*) take(128 * 4);
  float*  focf    = (float*) take(128 * 4);

  // 1. weight prep
  prep_weights<<<256, 256, 0, stream>>>(m1_Win, m1_Wout, m1_Wx, m1_Alog,
                                        Winbf, Woutbf, Wxpadbf, A1);
  // 2. directional sequences
  build_fop<<<(Mrows * DM1) / 256, 256, 0, stream>>>(fo1, fopbf);
  // 3. GEMM1: xz = fop @ Win^T   (16384 x 512, K=128), 4 N-tiles per wave
  wmma_gemm_bf16<4><<<dim3(Mrows / 16, 512 / 64), 32, 0, stream>>>(
      fopbf, Winbf, xz, Mrows, 512, DM1);
  // 4. conv + silu
  conv_silu<<<(Mrows * DI1) / 256, 256, 0, stream>>>(xz, m1_convw, m1_convb,
                                                     xa, xabf);
  // 5. GEMM2: xdbl = xa @ Wx^T   (16384 x 48, K=256), 3 N-tiles per wave
  wmma_gemm_bf16<3><<<dim3(Mrows / 16, 1), 32, 0, stream>>>(
      xabf, Wxpadbf, xdbl, Mrows, XDBLN, DI1);
  // 6. dt softplus
  dt_softplus<<<(Mrows * DI1) / 256, 256, 0, stream>>>(xdbl, m1_Wdt, m1_dtb, dt);
  // 7. selective scan + gating
  scan_gate<<<Bsz, 256, 0, stream>>>(dt, xa, xdbl, A1, m1_D, xz, ygbf);
  // 8. GEMM3: out1 = yg @ Wout^T (16384 x 128, K=256), 4 N-tiles per wave
  wmma_gemm_bf16<4><<<dim3(Mrows / 16, DM1 / 64), 32, 0, stream>>>(
      ygbf, Woutbf, out1, Mrows, DM1, DI1);
  // 9. direction sum + residual
  sum_residual<<<(Bsz * Cch * Lseq) / 256, 256, 0, stream>>>(out1, fo2, residual);
  // 10. channel means
  foc_mean<<<Bsz * Cch, 256, 0, stream>>>(residual, focm);
  // 11. tiny second Mamba
  mamba2_kernel<<<1, 32, 0, stream>>>(focm, m2_Win, m2_convw, m2_convb, m2_Wx,
                                      m2_Wdt, m2_dtb, m2_Alog, m2_D, m2_Wout,
                                      focf);
  // 12. final broadcast FMA
  final_kernel<<<(Bsz * Cch * Lseq) / 256, 256, 0, stream>>>(residual, focf, out);
}
